// LSTM_41025527611729
// MI455X (gfx1250) — compile-verified, matrix operand order checked by reading, not run on hardware
//
#include <hip/hip_runtime.h>
#include <hip/hip_bf16.h>
#include <stdint.h>

// ---------------------------------------------------------------------------
// Problem constants
// ---------------------------------------------------------------------------
#define BATCH   128
#define SEQ_T   512
#define IN_I    64
#define HID_H   256
#define G4H     1024      // 4*H gate columns
#define BT      16        // batch tile per workgroup (WMMA M)
#define NWG     (BATCH / BT)          // 8 workgroups
#define WAVES   16                    // waves per workgroup
#define BLK     (WAVES * 32)          // 512 threads

typedef __attribute__((ext_vector_type(16))) __bf16 v16bf;
typedef __attribute__((ext_vector_type(8)))  __bf16 v8bf;
typedef __attribute__((ext_vector_type(8)))  float  v8f;

// fast sigmoid / tanh via v_exp_f32
__device__ __forceinline__ float sigf(float x) { return 1.0f / (1.0f + __expf(-x)); }
__device__ __forceinline__ float tanhfast(float x) { return 2.0f * sigf(2.0f * x) - 1.0f; }
__device__ __forceinline__ float lrelu(float x) { return x > 0.0f ? x : 0.01f * x; }

// ---------------------------------------------------------------------------
// Prep: f32 -> bf16 copy of x
// ---------------------------------------------------------------------------
__global__ void cvt_bf16_kernel(const float* __restrict__ src,
                                unsigned short* __restrict__ dst, int n) {
    int i = blockIdx.x * blockDim.x + threadIdx.x;
    if (i < n) {
        __bf16 b = (__bf16)src[i];
        dst[i] = *(const unsigned short*)&b;
    }
}

// ---------------------------------------------------------------------------
// Prep: combined bias b_ih + b_hh
// ---------------------------------------------------------------------------
__global__ void bias_kernel(const float* __restrict__ bih,
                            const float* __restrict__ bhh,
                            float* __restrict__ bsum) {
    int i = blockIdx.x * blockDim.x + threadIdx.x;
    if (i < G4H) bsum[i] = bih[i] + bhh[i];
}

// ---------------------------------------------------------------------------
// Prep: pack [w_ih | w_hh] (f32, row-major (4H x in) / (4H x H)) into bf16
// WMMA B-fragment order.  Element index layout:
//   idx = (((nt*KT + kt)*32 + lane)*16) + e     (nt: 16-wide N tile, 64 total)
//   lane<16 -> n = nt*16 + lane,     K chunk offsets {0..7, 16..23}
//   lane>=16-> n = nt*16 + lane-16,  K chunk offsets {8..15, 24..31}
// so each lane's 16 bf16 fragment elements are 32 contiguous bytes.
// ---------------------------------------------------------------------------
__global__ void pack_w_kernel(const float* __restrict__ w_ih,
                              const float* __restrict__ w_hh,
                              unsigned short* __restrict__ wpack,
                              int in_sz, int KT, int total) {
    int idx = blockIdx.x * blockDim.x + threadIdx.x;
    if (idx >= total) return;
    int e16  = idx & 15;
    int t1   = idx >> 4;
    int lane = t1 & 31;
    int t2   = t1 >> 5;
    int kt   = t2 % KT;
    int nt   = t2 / KT;
    int k = kt * 32 + ((e16 >> 3) << 4) + ((lane >> 4) << 3) + (e16 & 7);
    int n = nt * 16 + (lane & 15);
    float w = (k < in_sz) ? w_ih[n * in_sz + k] : w_hh[n * HID_H + (k - in_sz)];
    __bf16 b = (__bf16)w;
    wpack[idx] = *(const unsigned short*)&b;
}

// ---------------------------------------------------------------------------
// Persistent LSTM layer kernel (IN_SZ is compile-time: 64 or 256).
//   grid = 8 WGs (one 16-batch tile each), block = 512 threads (16 waves).
//   Wave w owns hidden units [w*16, w*16+16): gate tiles i/f/g/o of 16x16.
//   z = [x_t | h_{t-1}] kept in LDS as bf16 with padded row stride.
//   Weights streamed from L2 in WMMA fragment order, register double-buffered
//   so each K-iteration's loads overlap the previous iteration's WMMAs.
// ---------------------------------------------------------------------------
template <int IN_SZ>
__global__ void __launch_bounds__(BLK)
lstm_layer_kernel(const unsigned short* __restrict__ xin,   // (B,T,IN_SZ) bf16
                  const unsigned short* __restrict__ wpack, // packed (K x 1024) bf16
                  const float* __restrict__ bsum,           // (1024) f32
                  unsigned short* __restrict__ hout) {      // (B,T,H) bf16
    constexpr int Ktot = IN_SZ + HID_H;     // 320 or 512 (multiple of 32)
    constexpr int KT   = Ktot >> 5;         // K tiles of 32
    constexpr int ZS   = Ktot + 8;          // padded LDS row stride (bf16 elems)
    constexpr int KSTEP = 32 * 16;          // packed elements per K-tile per N-tile

    __shared__ __attribute__((aligned(16))) unsigned short zbuf[BT * ZS];

    const int tid  = threadIdx.x;
    const int lane = tid & 31;
    const int wave = tid >> 5;
    const int b0   = blockIdx.x * BT;       // batch tile base
    const int hb   = wave * 16;             // wave's hidden-unit base
    const int lcol = lane & 15;             // N index within tile
    const int mrow = (lane >> 4) << 3;      // +8 row offset for hi half-wave

    // per-lane bias (broadcast along M inside each accumulator)
    const float bias_i = bsum[0 * HID_H + hb + lcol];
    const float bias_f = bsum[1 * HID_H + hb + lcol];
    const float bias_g = bsum[2 * HID_H + hb + lcol];
    const float bias_o = bsum[3 * HID_H + hb + lcol];

    // zero the z buffer (x cols overwritten each step; h cols start at h=0)
    for (int i = tid; i < (BT * ZS) >> 1; i += BLK)
        ((unsigned int*)zbuf)[i] = 0u;

    // cell state tile (16x16) in D layout: elem v <-> (m = v + mrow, j = hb+lcol)
    v8f cst;
#pragma unroll
    for (int v = 0; v < 8; ++v) cst[v] = 0.0f;

    // A-fragment LDS source addresses (per lane), fixed across t
    const int arow  = lane & 15;
    const int akoff = (lane >> 4) << 3;     // 0 or 8
    unsigned short* const zrow = &zbuf[arow * ZS];

    // per-gate packed-weight base pointers for this wave (kt = 0)
    const size_t fb = ((size_t)lane << 4);  // 32B fragment chunk per lane
    const unsigned short* const wbase0 = wpack + (size_t)((0 * 16 + wave) * KT) * KSTEP + fb;
    const unsigned short* const wbase1 = wpack + (size_t)((1 * 16 + wave) * KT) * KSTEP + fb;
    const unsigned short* const wbase2 = wpack + (size_t)((2 * 16 + wave) * KT) * KSTEP + fb;
    const unsigned short* const wbase3 = wpack + (size_t)((3 * 16 + wave) * KT) * KSTEP + fb;

    __syncthreads();

    constexpr int hpairs = IN_SZ >> 1;      // uint pairs per x row (pow2)

    for (int t = 0; t < SEQ_T; ++t) {
        // ---- stage x_t into z (cooperative, coalesced uint copies) ----
#pragma unroll
        for (int i = tid; i < BT * hpairs; i += BLK) {
            int m  = i >> (IN_SZ == 64 ? 5 : 7);
            int kp = i & (hpairs - 1);
            unsigned int val =
                *(const unsigned int*)&xin[(((size_t)(b0 + m) * SEQ_T) + t) * IN_SZ + 2 * kp];
            *(unsigned int*)&zbuf[m * ZS + 2 * kp] = val;
        }
        __syncthreads();   // x_t and h_{t-1} visible to every wave

        // ---- gates = [x_t | h] @ Wpack + bias  (WMMA, f32 accum) ----
        v8f acc_i, acc_f, acc_g, acc_o;
#pragma unroll
        for (int v = 0; v < 8; ++v) {
            acc_i[v] = bias_i; acc_f[v] = bias_f;
            acc_g[v] = bias_g; acc_o[v] = bias_o;
        }

        const unsigned short* p0 = wbase0;
        const unsigned short* p1 = wbase1;
        const unsigned short* p2 = wbase2;
        const unsigned short* p3 = wbase3;

        // preload kt=0 B fragments (one contiguous 32B chunk per lane per gate)
        v16bf bfi = *(const v16bf*)p0;
        v16bf bff = *(const v16bf*)p1;
        v16bf bfg = *(const v16bf*)p2;
        v16bf bfo = *(const v16bf*)p3;

        for (int kt = 0; kt < KT; ++kt) {
            // issue next K-tile's B loads first (register double buffer);
            // the tail read past the gate block lands in the 4KB pad.
            p0 += KSTEP; p1 += KSTEP; p2 += KSTEP; p3 += KSTEP;
            v16bf nfi = *(const v16bf*)p0;
            v16bf nff = *(const v16bf*)p1;
            v16bf nfg = *(const v16bf*)p2;
            v16bf nfo = *(const v16bf*)p3;
            __builtin_prefetch((const void*)(p0 + KSTEP), 0, 1);

            // A fragment: 2x 16B LDS loads (bank-conflict-free padded stride)
            union { v16bf v; v8bf h[2]; } A;
            A.h[0] = *(const v8bf*)&zrow[kt * 32 + akoff];
            A.h[1] = *(const v8bf*)&zrow[kt * 32 + akoff + 16];

            acc_i = __builtin_amdgcn_wmma_f32_16x16x32_bf16(false, A.v, false, bfi,
                                                            (short)0, acc_i, false, false);
            acc_f = __builtin_amdgcn_wmma_f32_16x16x32_bf16(false, A.v, false, bff,
                                                            (short)0, acc_f, false, false);
            acc_g = __builtin_amdgcn_wmma_f32_16x16x32_bf16(false, A.v, false, bfg,
                                                            (short)0, acc_g, false, false);
            acc_o = __builtin_amdgcn_wmma_f32_16x16x32_bf16(false, A.v, false, bfo,
                                                            (short)0, acc_o, false, false);
            bfi = nfi; bff = nff; bfg = nfg; bfo = nfo;
        }

        __syncthreads();   // everyone finished reading z before h is overwritten

        // ---- elementwise LSTM cell update + h writeback ----
#pragma unroll
        for (int v = 0; v < 8; ++v) {
            float iv = sigf(acc_i[v]);
            float fv = sigf(acc_f[v]);
            float gv = tanhfast(acc_g[v]);
            float ov = sigf(acc_o[v]);
            float c  = fv * cst[v] + iv * gv;
            cst[v]   = c;
            float h  = ov * tanhfast(c);
            __bf16 hb16 = (__bf16)h;
            unsigned short hu = *(const unsigned short*)&hb16;
            int m = v + mrow;
            zbuf[m * ZS + IN_SZ + hb + lcol] = hu;                        // LDS for t+1
            hout[(((size_t)(b0 + m) * SEQ_T) + t) * HID_H + hb + lcol] = hu; // next layer
        }
        __syncthreads();   // h_t complete before next iteration reads z
    }
}

// ---------------------------------------------------------------------------
// MLP head: last_h -> fc1(256->128) -> fc2(128->32) -> fc3(32->2) -> softmax
// one block per batch element
// ---------------------------------------------------------------------------
__global__ void __launch_bounds__(128)
head_kernel(const unsigned short* __restrict__ hseq,
            const float* __restrict__ w1, const float* __restrict__ b1,
            const float* __restrict__ w2, const float* __restrict__ b2,
            const float* __restrict__ w3, const float* __restrict__ b3,
            float* __restrict__ out) {
    __shared__ float hv[HID_H];
    __shared__ float y1[128];
    __shared__ float y2[32];
    __shared__ float y3[2];
    const int b = blockIdx.x, tid = threadIdx.x;

    for (int i = tid; i < HID_H; i += 128) {
        unsigned short u = hseq[(((size_t)b * SEQ_T) + (SEQ_T - 1)) * HID_H + i];
        hv[i] = (float)(*(const __bf16*)&u);
    }
    __syncthreads();

    {
        float a = b1[tid];
        for (int k = 0; k < HID_H; ++k) a += hv[k] * w1[tid * HID_H + k];
        y1[tid] = lrelu(a);
    }
    __syncthreads();
    if (tid < 32) {
        float a = b2[tid];
        for (int k = 0; k < 128; ++k) a += y1[k] * w2[tid * 128 + k];
        y2[tid] = lrelu(a);
    }
    __syncthreads();
    if (tid < 2) {
        float a = b3[tid];
        for (int k = 0; k < 32; ++k) a += y2[k] * w3[tid * 32 + k];
        y3[tid] = lrelu(a);
    }
    __syncthreads();
    if (tid == 0) {
        float m  = fmaxf(y3[0], y3[1]);
        float e0 = __expf(y3[0] - m), e1 = __expf(y3[1] - m);
        float s  = e0 + e1;
        out[b * 2 + 0] = e0 / s;
        out[b * 2 + 1] = e1 / s;
    }
}

// ---------------------------------------------------------------------------
// Host launcher
// ---------------------------------------------------------------------------
extern "C" void kernel_launch(void* const* d_in, const int* in_sizes, int n_in,
                              void* d_out, int out_size, void* d_ws, size_t ws_size,
                              hipStream_t stream) {
    const float* x = (const float*)d_in[0];
    const float* w_ih[3] = {(const float*)d_in[1], (const float*)d_in[5], (const float*)d_in[9]};
    const float* w_hh[3] = {(const float*)d_in[2], (const float*)d_in[6], (const float*)d_in[10]};
    const float* b_ih[3] = {(const float*)d_in[3], (const float*)d_in[7], (const float*)d_in[11]};
    const float* b_hh[3] = {(const float*)d_in[4], (const float*)d_in[8], (const float*)d_in[12]};
    const float* fc1_w = (const float*)d_in[13];
    const float* fc1_b = (const float*)d_in[14];
    const float* fc2_w = (const float*)d_in[15];
    const float* fc2_b = (const float*)d_in[16];
    const float* fc3_w = (const float*)d_in[17];
    const float* fc3_b = (const float*)d_in[18];
    float* out = (float*)d_out;

    // workspace carve-up (bytes), all 256-aligned; wpack buffers get a 4KB
    // tail pad so the pipelined preload may harmlessly read one K-tile past.
    char* ws = (char*)d_ws;
    unsigned short* xbf   = (unsigned short*)(ws);                       // 8 MB
    unsigned short* hA    = (unsigned short*)(ws + 8388608ull);          // 32 MB
    unsigned short* hB    = (unsigned short*)(ws + 41943040ull);         // 32 MB
    unsigned short* wp0   = (unsigned short*)(ws + 75497472ull);         // 640 KB + pad
    unsigned short* wp1   = (unsigned short*)(ws + 76157184ull);         // 1 MB + pad
    unsigned short* wp2   = (unsigned short*)(ws + 77209856ull);         // 1 MB + pad
    float*          bs0   = (float*)(ws + 78262528ull);                  // 4 KB each
    float*          bs1   = (float*)(ws + 78266624ull);
    float*          bs2   = (float*)(ws + 78270720ull);
    unsigned short* wp[3] = {wp0, wp1, wp2};
    float*          bs[3] = {bs0, bs1, bs2};

    // ---- prep ----
    {
        int n = BATCH * SEQ_T * IN_I;
        cvt_bf16_kernel<<<(n + 255) / 256, 256, 0, stream>>>(x, xbf, n);
    }
    for (int l = 0; l < 3; ++l) {
        int in_sz = (l == 0) ? IN_I : HID_H;
        int Ktot  = in_sz + HID_H;
        int KT    = Ktot / 32;
        int total = Ktot * G4H;
        bias_kernel<<<4, 256, 0, stream>>>(b_ih[l], b_hh[l], bs[l]);
        pack_w_kernel<<<(total + 255) / 256, 256, 0, stream>>>(
            w_ih[l], w_hh[l], wp[l], in_sz, KT, total);
    }

    // ---- persistent LSTM layers (serial over layers, 8 WGs each) ----
    lstm_layer_kernel<IN_I><<<NWG, BLK, 0, stream>>>(xbf, wp0, bs0, hA);
    lstm_layer_kernel<HID_H><<<NWG, BLK, 0, stream>>>(hA, wp1, bs1, hB);
    lstm_layer_kernel<HID_H><<<NWG, BLK, 0, stream>>>(hB, wp2, bs2, hA);

    // ---- head ----
    head_kernel<<<BATCH, 128, 0, stream>>>(hA, fc1_w, fc1_b, fc2_w, fc2_b,
                                           fc3_w, fc3_b, out);
    (void)in_sizes; (void)n_in; (void)out_size; (void)ws_size;
}
